// Block_5188320494404
// MI455X (gfx1250) — compile-verified, matrix operand order checked by reading, not run on hardware
//
#include <hip/hip_runtime.h>
#include <hip/hip_bf16.h>
#include <math.h>

typedef __attribute__((ext_vector_type(16))) __bf16 bf16x16;
typedef __attribute__((ext_vector_type(8)))  __bf16 bf16x8;
typedef __attribute__((ext_vector_type(8)))  float  f32x8;

constexpr int Bc = 32, Tc = 512, Cc = 512, Hc = 8, Dc = 64;
constexpr int Ec = 4, Fc = 2048, WINc = 128;
constexpr int Ntok = Bc * Tc;              // 16384 tokens

// Flip to 0 if the assembler rejects the async-to-LDS mnemonics.
#define USE_ASYNC_LDS 1

// ---------------------------------------------------------------------------
// WMMA helpers (CDNA5 16x16x32 bf16, f32 accumulate)
// A-layout (16-bit, 16x32): lane<16 -> row=lane, K = k0+{0..7,16..23};
//                           lane>=16 -> row=lane-16, K = k0+{8..15,24..31}.
// B mirrored (lane indexes the N column, same K packing) -> load from B^T.
// C/D layout: VGPR r -> (M = r + 8*(lane>=16), N = lane&15).
// ---------------------------------------------------------------------------
__device__ __forceinline__ f32x8 zero8() {
  f32x8 z = {0.f, 0.f, 0.f, 0.f, 0.f, 0.f, 0.f, 0.f};
  return z;
}

__device__ __forceinline__ bf16x16 ld_ab(const __bf16* base, int ld, int k0, unsigned lane) {
  unsigned r  = lane & 15u;
  unsigned hi = (lane >> 4) & 1u;
  const __bf16* p = base + (size_t)r * ld + (unsigned)k0 + hi * 8u;
  bf16x8 lo = *(const bf16x8*)(p);
  bf16x8 hv = *(const bf16x8*)(p + 16);
  bf16x16 o;
#pragma unroll
  for (int i = 0; i < 8; ++i) { o[i] = lo[i]; o[i + 8] = hv[i]; }
  return o;
}

__device__ __forceinline__ f32x8 wmma_bf16(bf16x16 a, bf16x16 b, f32x8 c) {
  return __builtin_amdgcn_wmma_f32_16x16x32_bf16(false, a, false, b, (short)0, c, false, false);
}

__device__ __forceinline__ f32x8 gemm_tile(const __bf16* A, int lda,
                                           const __bf16* Bt, int ldb,
                                           int K, unsigned lane) {
  f32x8 c = zero8();
  for (int k0 = 0; k0 < K; k0 += 32) {
    bf16x16 a = ld_ab(A, lda, k0, lane);
    bf16x16 b = ld_ab(Bt, ldb, k0, lane);
    c = wmma_bf16(a, b, c);
  }
  return c;
}

// async global -> LDS 16B copy (CDNA5 Tensor/async data path), with fallback
__device__ __forceinline__ void copy16_to_lds(unsigned lds_byte_off, const __bf16* gsrc) {
#if USE_ASYNC_LDS
  unsigned long long ga = (unsigned long long)(const void*)gsrc;
  asm volatile("global_load_async_to_lds_b128 %0, %1, off"
               :: "v"(lds_byte_off), "v"(ga) : "memory");
#else
  (void)lds_byte_off;
#endif
}

// ---------------------------------------------------------------------------
// Weight convert + transpose: out[e][n][k] = (bf16) in[e][k][n]
// ---------------------------------------------------------------------------
__global__ void cvt_transpose_kernel(const float* __restrict__ in,
                                     __bf16* __restrict__ out,
                                     int Eb, int Kb, int Nb) {
  long i = (long)blockIdx.x * blockDim.x + threadIdx.x;
  long total = (long)Eb * Kb * Nb;
  if (i >= total) return;
  int n = (int)(i % Nb);
  long t = i / Nb;
  int k = (int)(t % Kb);
  int e = (int)(t / Kb);
  out[((long)e * Nb + n) * Kb + k] = (__bf16)in[((long)e * Kb + k) * Nb + n];
}

// ---------------------------------------------------------------------------
// LayerNorm -> bf16 (wave per token)
// ---------------------------------------------------------------------------
__global__ void __launch_bounds__(256) ln_bf16_kernel(const float* __restrict__ x,
                                                      const float* __restrict__ g,
                                                      const float* __restrict__ b,
                                                      __bf16* __restrict__ out) {
  int lane = threadIdx.x & 31, wv = threadIdx.x >> 5;
  int n = blockIdx.x * 8 + wv;
  const float* xr = x + (size_t)n * Cc;
  float s = 0.f, s2 = 0.f;
  for (int c = lane; c < Cc; c += 32) { float v = xr[c]; s += v; s2 += v * v; }
  for (int o = 16; o > 0; o >>= 1) { s += __shfl_xor(s, o, 32); s2 += __shfl_xor(s2, o, 32); }
  float m = s / Cc, var = s2 / Cc - m * m;
  float r = rsqrtf(var + 1e-5f);
  __bf16* op = out + (size_t)n * Cc;
  for (int c = lane; c < Cc; c += 32) op[c] = (__bf16)((xr[c] - m) * r * g[c] + b[c]);
}

// ---------------------------------------------------------------------------
// QKV projection: per-wave 16x16 tile of q/k/v[b,h,t,d]
// ---------------------------------------------------------------------------
__global__ void __launch_bounds__(256) qkv_kernel(const __bf16* __restrict__ h1,
                                                  const __bf16* __restrict__ WqT,
                                                  const __bf16* __restrict__ WkT,
                                                  const __bf16* __restrict__ WvT,
                                                  __bf16* __restrict__ q,
                                                  __bf16* __restrict__ k,
                                                  __bf16* __restrict__ v) {
  unsigned lane = threadIdx.x & 31u;
  int wid = blockIdx.x * 8 + (threadIdx.x >> 5);
  int mt = wid / 96;
  int rem = wid % 96;
  int h = rem / 12; rem %= 12;
  int which = rem / 4;
  int dt = rem % 4;
  const __bf16* Wt  = which == 0 ? WqT : (which == 1 ? WkT : WvT);
  __bf16*       outp = which == 0 ? q  : (which == 1 ? k  : v);

  f32x8 c = gemm_tile(h1 + (size_t)mt * 16 * Cc, Cc,
                      Wt + ((size_t)h * Dc + dt * 16) * Cc, Cc, Cc, lane);

  int bidx = mt / (Tc / 16);
  int tl = (mt % (Tc / 16)) * 16;
  unsigned col = lane & 15u;
  unsigned rb = ((lane >> 4) & 1u) * 8u;
  __bf16* ob = outp + (((size_t)bidx * Hc + h) * Tc + tl) * Dc + dt * 16 + col;
#pragma unroll
  for (int r = 0; r < 8; ++r) ob[(size_t)(rb + r) * Dc] = (__bf16)c[r];
}

// ---------------------------------------------------------------------------
// Sliding-window flash attention: one wave per (b,h,16-query tile)
// ---------------------------------------------------------------------------
__global__ void __launch_bounds__(256) attn_kernel(const __bf16* __restrict__ q,
                                                   const __bf16* __restrict__ k,
                                                   const __bf16* __restrict__ v,
                                                   __bf16* __restrict__ obf) {
  __shared__ __bf16 ldsP[8][16][32];
  unsigned lane = threadIdx.x & 31u;
  int wv = threadIdx.x >> 5;
  int wid = blockIdx.x * 8 + wv;
  int b = wid / (Hc * (Tc / 16));
  int rem = wid % (Hc * (Tc / 16));
  int h = rem / (Tc / 16);
  int qt = rem % (Tc / 16);
  int t0 = qt * 16;

  const __bf16* qb = q + (((size_t)b * Hc + h) * Tc + t0) * Dc;
  const __bf16* kb = k + ((size_t)b * Hc + h) * Tc * Dc;
  const __bf16* vb = v + ((size_t)b * Hc + h) * Tc * Dc;

  bf16x16 aq0 = ld_ab(qb, Dc, 0, lane);
  bf16x16 aq1 = ld_ab(qb, Dc, 32, lane);

  f32x8 acc[4];
#pragma unroll
  for (int t = 0; t < 4; ++t) acc[t] = zero8();
  float m_[8], l_[8];
#pragma unroll
  for (int r = 0; r < 8; ++r) { m_[r] = -1e30f; l_[r] = 0.f; }

  unsigned col = lane & 15u;
  unsigned rb = ((lane >> 4) & 1u) * 8u;
  int s_lo = t0 - (WINc - 1); if (s_lo < 0) s_lo = 0;
  int s_start = (s_lo / 32) * 32;
  int nsteps = (t0 + 16 - s_start + 31) / 32;
  const float rs = 0.044194173824159216f;  // 1/sqrt(C) (reference scales by C)

  for (int st = 0; st < nsteps; ++st) {
    int s0 = s_start + st * 32;
    f32x8 sc[2];
#pragma unroll
    for (int sub = 0; sub < 2; ++sub) {
      f32x8 cacc = zero8();
      bf16x16 bk0 = ld_ab(kb + (size_t)(s0 + sub * 16) * Dc, Dc, 0, lane);
      bf16x16 bk1 = ld_ab(kb + (size_t)(s0 + sub * 16) * Dc, Dc, 32, lane);
      cacc = wmma_bf16(aq0, bk0, cacc);
      cacc = wmma_bf16(aq1, bk1, cacc);
      sc[sub] = cacc;
    }
    float scale[8];
#pragma unroll
    for (int r = 0; r < 8; ++r) {
      int row = t0 + (int)rb + r;
      float mt_ = -1e30f;
#pragma unroll
      for (int sub = 0; sub < 2; ++sub) {
        int s = s0 + sub * 16 + (int)col;
        float sv = sc[sub][r] * rs;
        bool ok = (row >= s) && (row - s < WINc);
        sv = ok ? sv : -1e30f;
        sc[sub][r] = sv;
        mt_ = fmaxf(mt_, sv);
      }
      for (int o = 1; o < 16; o <<= 1) mt_ = fmaxf(mt_, __shfl_xor(mt_, o, 32));
      float mn = fmaxf(m_[r], mt_);
      scale[r] = expf(m_[r] - mn);
      float ps = 0.f;
#pragma unroll
      for (int sub = 0; sub < 2; ++sub) {
        float sv = sc[sub][r];
        float pv = (sv > -0.9e30f) ? expf(sv - mn) : 0.f;
        ps += pv;
        ldsP[wv][rb + r][sub * 16 + col] = (__bf16)pv;
      }
      for (int o = 1; o < 16; o <<= 1) ps += __shfl_xor(ps, o, 32);
      l_[r] = l_[r] * scale[r] + ps;
      m_[r] = mn;
    }
#pragma unroll
    for (int t = 0; t < 4; ++t)
#pragma unroll
      for (int r = 0; r < 8; ++r) acc[t][r] *= scale[r];

    asm volatile("s_wait_dscnt 0" ::: "memory");
    bf16x16 ap = ld_ab(&ldsP[wv][0][0], 32, 0, lane);
#pragma unroll
    for (int dt = 0; dt < 4; ++dt) {
      bf16x16 bv;
      const __bf16* vp = vb + ((size_t)s0 + ((lane >= 16) ? 8 : 0)) * Dc + dt * 16 + col;
#pragma unroll
      for (int j = 0; j < 8; ++j) {
        bv[j]     = vp[(size_t)j * Dc];
        bv[j + 8] = vp[(size_t)(j + 16) * Dc];
      }
      acc[dt] = wmma_bf16(ap, bv, acc[dt]);
    }
  }

  float inv[8];
#pragma unroll
  for (int r = 0; r < 8; ++r) inv[r] = 1.f / l_[r];
#pragma unroll
  for (int dt = 0; dt < 4; ++dt)
#pragma unroll
    for (int r = 0; r < 8; ++r) {
      size_t n = (size_t)b * Tc + t0 + rb + r;
      obf[n * Cc + h * Dc + dt * 16 + col] = (__bf16)(acc[dt][r] * inv[r]);
    }
}

// ---------------------------------------------------------------------------
// Output projection + residual: x2 = x + o @ Wp + bp   (fp32 out)
// ---------------------------------------------------------------------------
__global__ void __launch_bounds__(256) oproj_kernel(const __bf16* __restrict__ obf,
                                                    const __bf16* __restrict__ WpT,
                                                    const float* __restrict__ x,
                                                    const float* __restrict__ bp,
                                                    float* __restrict__ x2) {
  unsigned lane = threadIdx.x & 31u;
  int wid = blockIdx.x * 8 + (threadIdx.x >> 5);
  int mt = wid / (Cc / 16);
  int ct = wid % (Cc / 16);
  f32x8 c = gemm_tile(obf + (size_t)mt * 16 * Cc, Cc,
                      WpT + (size_t)ct * 16 * Cc, Cc, Cc, lane);
  unsigned col = lane & 15u;
  unsigned rb = ((lane >> 4) & 1u) * 8u;
  int n0 = mt * 16;
  int c0 = ct * 16 + (int)col;
  float bpv = bp[c0];
#pragma unroll
  for (int r = 0; r < 8; ++r) {
    size_t idx = (size_t)(n0 + rb + r) * Cc + c0;
    x2[idx] = x[idx] + c[r] + bpv;
  }
}

// ---------------------------------------------------------------------------
// LN2 + gating + top-2 softmax routing weights
// ---------------------------------------------------------------------------
__global__ void __launch_bounds__(256) ln2_gate_kernel(const float* __restrict__ x2,
                                                       const float* __restrict__ g,
                                                       const float* __restrict__ b,
                                                       const float* __restrict__ Wg,
                                                       const float* __restrict__ bg,
                                                       __bf16* __restrict__ h2,
                                                       float* __restrict__ wout) {
  int lane = threadIdx.x & 31, wv = threadIdx.x >> 5;
  int n = blockIdx.x * 8 + wv;
  const float* xr = x2 + (size_t)n * Cc;
  float s = 0.f, s2 = 0.f;
  for (int c = lane; c < Cc; c += 32) { float v = xr[c]; s += v; s2 += v * v; }
  for (int o = 16; o > 0; o >>= 1) { s += __shfl_xor(s, o, 32); s2 += __shfl_xor(s2, o, 32); }
  float m = s / Cc, var = s2 / Cc - m * m;
  float r = rsqrtf(var + 1e-5f);
  float ge[4] = {0.f, 0.f, 0.f, 0.f};
  for (int c = lane; c < Cc; c += 32) {
    float hv = (xr[c] - m) * r * g[c] + b[c];
    h2[(size_t)n * Cc + c] = (__bf16)hv;
#pragma unroll
    for (int e = 0; e < 4; ++e) ge[e] += hv * Wg[(size_t)c * Ec + e];
  }
  for (int o = 16; o > 0; o >>= 1)
#pragma unroll
    for (int e = 0; e < 4; ++e) ge[e] += __shfl_xor(ge[e], o, 32);
  if (lane == 0) {
#pragma unroll
    for (int e = 0; e < 4; ++e) ge[e] += bg[e];
    int i0 = 0; float v0 = ge[0];
    for (int e = 1; e < 4; ++e) if (ge[e] > v0) { v0 = ge[e]; i0 = e; }
    int i1 = (i0 == 0) ? 1 : 0; float v1 = -1e30f;
    for (int e = 0; e < 4; ++e) if (e != i0 && ge[e] > v1) { v1 = ge[e]; i1 = e; }
    float e1 = expf(v1 - v0);
    float inv = 1.f / (1.f + e1);
    float o4[4] = {0.f, 0.f, 0.f, 0.f};
    o4[i0] += inv;
    o4[i1] += e1 * inv;
#pragma unroll
    for (int e = 0; e < 4; ++e) wout[(size_t)n * 4 + e] = o4[e];
  }
}

// ---------------------------------------------------------------------------
// Fused MoE FFN (+ residual, final output). M-tile = 64 tokens per block
// (4x less expert-weight traffic from L2 than a 16-token tile).
// Routing weight folded into hidden activations, so one accumulator
// (16 tiles = 128 VGPRs/lane) carries the sum over experts AND f-chunks.
// Token activations staged once to LDS via async global->LDS loads.
// wave w: m-subtile = w&3 (16 rows), output cols = (w>>2)*256 .. +255.
// ---------------------------------------------------------------------------
__global__ void __launch_bounds__(256, 1) moe_kernel(const __bf16* __restrict__ h2,
                                                     const __bf16* __restrict__ W1T,
                                                     const float* __restrict__ b1,
                                                     const __bf16* __restrict__ W2T,
                                                     const float* __restrict__ b2,
                                                     const float* __restrict__ wgt,
                                                     const float* __restrict__ x2,
                                                     float* __restrict__ out) {
  __shared__ __bf16 h2s[64][520];      // 64 tokens x 512 (+8 pad: bank-spread, 16B rows)
  __shared__ __bf16 hidbuf[64][136];   // 64 tokens x 128 f (+8 pad)
  __shared__ float  wtile[64][4];
  unsigned lane = threadIdx.x & 31u;
  int wv = threadIdx.x >> 5;
  int n0 = blockIdx.x * 64;

  // stage routing weights (exactly 256 = 64x4 elements)
  wtile[threadIdx.x >> 2][threadIdx.x & 3] =
      wgt[(size_t)(n0 + (threadIdx.x >> 2)) * 4 + (threadIdx.x & 3)];

  // stage the 64x512 bf16 activation tile into LDS (async data path)
  {
    unsigned ldsbase = (unsigned)(size_t)(void*)&h2s[0][0];
#pragma unroll
    for (int i = 0; i < 16; ++i) {
      unsigned u = threadIdx.x + i * 256;   // 4096 16B units total
      unsigned row = u >> 6, cu = u & 63u;  // 64 units per row
      const __bf16* g = h2 + (size_t)(n0 + row) * Cc + cu * 8u;
      unsigned loff = ldsbase + (row * 520u + cu * 8u) * 2u;
#if USE_ASYNC_LDS
      copy16_to_lds(loff, g);
#else
      *(bf16x8*)((__bf16*)&h2s[0][0] + row * 520u + cu * 8u) = *(const bf16x8*)g;
      (void)loff;
#endif
    }
  }
#if USE_ASYNC_LDS
  asm volatile("s_wait_asynccnt 0" ::: "memory");
#endif
  __syncthreads();

  unsigned col = lane & 15u;
  unsigned rb = ((lane >> 4) & 1u) * 8u;
  int msub = wv & 3;          // which 16-token row group
  int grp  = wv >> 2;         // 0/1: f-halves in GEMM1, 256-col halves in GEMM2
  const __bf16* Arow = &h2s[msub * 16][0];

  f32x8 outacc[16];
#pragma unroll
  for (int t = 0; t < 16; ++t) outacc[t] = zero8();

  for (int e = 0; e < Ec; ++e) {
    const __bf16* W1e = W1T + (size_t)e * Fc * Cc;  // [f][c]
    const __bf16* W2e = W2T + (size_t)e * Cc * Fc;  // [c][f]
    const float* b1e = b1 + (size_t)e * Fc;

    for (int fc = 0; fc < Fc / 128; ++fc) {
      int f0 = fc * 128;
      // ---- GEMM1: this wave computes hid rows msub, f cols grp*64..+63 ----
      f32x8 hacc[4];
#pragma unroll
      for (int t = 0; t < 4; ++t) hacc[t] = zero8();
      for (int kc = 0; kc < 16; ++kc) {
        bf16x16 a = ld_ab(Arow, 520, kc * 32, lane);
#pragma unroll
        for (int t = 0; t < 4; ++t) {
          bf16x16 bw = ld_ab(W1e + (size_t)(f0 + grp * 64 + t * 16) * Cc, Cc, kc * 32, lane);
          hacc[t] = wmma_bf16(a, bw, hacc[t]);
        }
      }
#pragma unroll
      for (int t = 0; t < 4; ++t) {
        float bv = b1e[f0 + grp * 64 + t * 16 + col];
#pragma unroll
        for (int r = 0; r < 8; ++r) {
          int row = msub * 16 + (int)rb + r;
          float hv = hacc[t][r] + bv;
          hv = hv > 0.f ? hv : 0.f;
          hv *= wtile[row][e];               // fold routing weight
          hidbuf[row][grp * 64 + t * 16 + col] = (__bf16)hv;
        }
      }
      __syncthreads();
      if (fc + 1 < Fc / 128) {
        __builtin_prefetch(W1e + (size_t)(f0 + 128 + grp * 64) * Cc, 0, 1);
        __builtin_prefetch(W2e + (size_t)(grp * 256 + (lane & 7) * 32) * Fc + f0 + 128, 0, 1);
      }
      // ---- GEMM2: out[msub rows, grp*256..+255] += hid[.,128] @ W2 ----
#pragma unroll
      for (int kc = 0; kc < 4; ++kc) {
        bf16x16 a2 = ld_ab(&hidbuf[msub * 16][0], 136, kc * 32, lane);
#pragma unroll
        for (int t = 0; t < 16; ++t) {
          bf16x16 b2v = ld_ab(W2e + (size_t)(grp * 256 + t * 16) * Fc + f0, Fc, kc * 32, lane);
          outacc[t] = wmma_bf16(a2, b2v, outacc[t]);
        }
      }
      __syncthreads();
    }
  }

  // epilogue: residual + routed bias  out = x2 + acc + sum_e w[n,e]*b2[e][c]
#pragma unroll
  for (int t = 0; t < 16; ++t) {
    int c0 = grp * 256 + t * 16 + (int)col;
    float bb[4];
#pragma unroll
    for (int e = 0; e < 4; ++e) bb[e] = b2[(size_t)e * Cc + c0];
#pragma unroll
    for (int r = 0; r < 8; ++r) {
      int lrow = msub * 16 + (int)rb + r;
      float bias = 0.f;
#pragma unroll
      for (int e = 0; e < 4; ++e) bias += wtile[lrow][e] * bb[e];
      size_t idx = (size_t)(n0 + lrow) * Cc + c0;
      out[idx] = x2[idx] + outacc[t][r] + bias;
    }
  }
}

// ---------------------------------------------------------------------------
// Host launcher
// ---------------------------------------------------------------------------
extern "C" void kernel_launch(void* const* d_in, const int* in_sizes, int n_in,
                              void* d_out, int out_size, void* d_ws, size_t ws_size,
                              hipStream_t stream) {
  const float* x    = (const float*)d_in[0];
  const float* ln1g = (const float*)d_in[1];
  const float* ln1b = (const float*)d_in[2];
  const float* Wq   = (const float*)d_in[3];
  const float* Wk   = (const float*)d_in[4];
  const float* Wv   = (const float*)d_in[5];
  const float* Wp   = (const float*)d_in[6];
  const float* bp   = (const float*)d_in[7];
  const float* ln2g = (const float*)d_in[8];
  const float* ln2b = (const float*)d_in[9];
  const float* Wg   = (const float*)d_in[10];
  const float* bg   = (const float*)d_in[11];
  const float* W1   = (const float*)d_in[12];
  const float* b1   = (const float*)d_in[13];
  const float* W2   = (const float*)d_in[14];
  const float* b2   = (const float*)d_in[15];
  float* out = (float*)d_out;

  char* p = (char*)d_ws;
  auto alloc = [&](size_t bytes) -> void* {
    void* r = (void*)p;
    p += (bytes + 255) & ~(size_t)255;
    return r;
  };
  __bf16* h1bf = (__bf16*)alloc((size_t)Ntok * Cc * 2);
  __bf16* qbf  = (__bf16*)alloc((size_t)Bc * Hc * Tc * Dc * 2);
  __bf16* kbf  = (__bf16*)alloc((size_t)Bc * Hc * Tc * Dc * 2);
  __bf16* vbf  = (__bf16*)alloc((size_t)Bc * Hc * Tc * Dc * 2);
  __bf16* obf  = (__bf16*)alloc((size_t)Ntok * Cc * 2);
  float*  x2   = (float*)alloc((size_t)Ntok * Cc * 4);
  __bf16* h2bf = (__bf16*)alloc((size_t)Ntok * Cc * 2);
  float*  wgt  = (float*)alloc((size_t)Ntok * Ec * 4);
  __bf16* WqT  = (__bf16*)alloc((size_t)Hc * Cc * Dc * 2);
  __bf16* WkT  = (__bf16*)alloc((size_t)Hc * Cc * Dc * 2);
  __bf16* WvT  = (__bf16*)alloc((size_t)Hc * Cc * Dc * 2);
  __bf16* WpT  = (__bf16*)alloc((size_t)Cc * Cc * 2);
  __bf16* W1T  = (__bf16*)alloc((size_t)Ec * Cc * Fc * 2);
  __bf16* W2T  = (__bf16*)alloc((size_t)Ec * Fc * Cc * 2);

  auto cvt = [&](const float* in, __bf16* o, int Eb, int Kb, int Nb) {
    long total = (long)Eb * Kb * Nb;
    int blocks = (int)((total + 255) / 256);
    cvt_transpose_kernel<<<blocks, 256, 0, stream>>>(in, o, Eb, Kb, Nb);
  };
  cvt(Wq, WqT, Hc, Cc, Dc);
  cvt(Wk, WkT, Hc, Cc, Dc);
  cvt(Wv, WvT, Hc, Cc, Dc);
  cvt(Wp, WpT, 1, Cc, Cc);
  cvt(W1, W1T, Ec, Cc, Fc);
  cvt(W2, W2T, Ec, Fc, Cc);

  ln_bf16_kernel<<<Ntok / 8, 256, 0, stream>>>(x, ln1g, ln1b, h1bf);

  qkv_kernel<<<(Ntok / 16) * Hc * 3 * (Dc / 16) / 8, 256, 0, stream>>>(
      h1bf, WqT, WkT, WvT, qbf, kbf, vbf);

  attn_kernel<<<Bc * Hc * (Tc / 16) / 8, 256, 0, stream>>>(qbf, kbf, vbf, obf);

  oproj_kernel<<<(Ntok / 16) * (Cc / 16) / 8, 256, 0, stream>>>(obf, WpT, x, bp, x2);

  ln2_gate_kernel<<<Ntok / 8, 256, 0, stream>>>(x2, ln2g, ln2b, Wg, bg, h2bf, wgt);

  moe_kernel<<<Ntok / 64, 256, 0, stream>>>(h2bf, W1T, b1, W2T, b2, wgt, x2, out);

  (void)in_sizes; (void)n_in; (void)out_size; (void)ws_size;
}
